// AnatomicalRelationshipModule_73246372266053
// MI455X (gfx1250) — matrix operand order repaired and checked by reading, not compiled, not run on hardware
//
#include <hip/hip_runtime.h>
#include <hip/hip_bf16.h>
#include <math.h>

// ---------------------------------------------------------------------------
// Types for CDNA5 WMMA (wave32): v_wmma_f32_16x16x32_bf16
// ---------------------------------------------------------------------------
typedef __bf16 bf16_t;
typedef bf16_t bf16x16 __attribute__((ext_vector_type(16)));
typedef bf16_t bf16x8  __attribute__((ext_vector_type(8)));
typedef float  v8f     __attribute__((ext_vector_type(8)));

#define HID   256
#define ROWS  32          // rows per block tile (2 M-tiles of 16)
#define NQ    1000
#define NB    64
#define BQ    (NB * NQ)   // 64000 flattened rows
#define LOGIT_THR 0.8472978603872034f   // log(0.7/0.3)  <=> sigmoid(x) > 0.7

__device__ __forceinline__ float sigmf(float x) { return 1.0f / (1.0f + expf(-x)); }
__device__ __forceinline__ float inv_sigm(float x) {
  x = fminf(fmaxf(x, 0.0f), 1.0f);
  float a = fmaxf(x, 1e-5f);
  float b = fmaxf(1.0f - x, 1e-5f);
  return logf(a / b);
}

// ---------------------------------------------------------------------------
// Kernel 0: W[k][n] (f32) -> WT[n][k] (bf16), 256x256
// ---------------------------------------------------------------------------
__global__ __launch_bounds__(256) void convert_transpose(const float* __restrict__ W,
                                                         bf16_t* __restrict__ WT) {
  int k = blockIdx.x;     // 0..255
  int n = threadIdx.x;    // 0..255
  WT[n * HID + k] = (bf16_t)W[k * HID + n];
}

// ---------------------------------------------------------------------------
// Kernel 1: candidate masks via pairwise IoU (writes cand_mask as 0/1 floats)
// grid = (4, 64): x = i-chunk, y = batch. 256 threads, one i per thread.
// ---------------------------------------------------------------------------
__global__ __launch_bounds__(256) void masks_kernel(const float* __restrict__ logits,
                                                    const float* __restrict__ boxes,
                                                    float* __restrict__ outMask) {
  __shared__ float jb[256][4];
  __shared__ unsigned char aL[256];
  __shared__ unsigned char aR[256];

  const int t = threadIdx.x;
  const int b = blockIdx.y;
  const int i = blockIdx.x * 256 + t;
  const bool vi = (i < NQ);

  float cx = 0.f, cy = 0.f, wd = 0.f, ht = 0.f;
  bool lf = false, rf = false;
  if (vi) {
    const float* lg = logits + ((long)b * NQ + i) * 6;
    lf = lg[0] > LOGIT_THR;   // left fold  (class 0)
    rf = lg[4] > LOGIT_THR;   // right fold (class 4)
    const float* bx = boxes + ((long)b * NQ + i) * 4;
    cx = bx[0]; cy = bx[1]; wd = bx[2]; ht = bx[3];
  }
  const float x1 = cx - 0.5f * wd, y1 = cy - 0.5f * ht;
  const float x2 = cx + 0.5f * wd, y2 = cy + 0.5f * ht;
  const float ai = wd * ht;

  bool hasL = false, hasR = false;
  for (int j0 = 0; j0 < NQ; j0 += 256) {
    const int jj = j0 + t;
    if (jj < NQ) {
      const float* bx = boxes + ((long)b * NQ + jj) * 4;
      jb[t][0] = bx[0]; jb[t][1] = bx[1]; jb[t][2] = bx[2]; jb[t][3] = bx[3];
      const float* lg = logits + ((long)b * NQ + jj) * 6;
      aL[t] = lg[1] > LOGIT_THR;  // left ary  (class 1)
      aR[t] = lg[5] > LOGIT_THR;  // right ary (class 5)
    } else {
      aL[t] = 0; aR[t] = 0;
      jb[t][0] = 0.f; jb[t][1] = 0.f; jb[t][2] = 0.f; jb[t][3] = 0.f;
    }
    __syncthreads();
    if (vi && (lf || rf)) {
      const int cnt = (NQ - j0) < 256 ? (NQ - j0) : 256;
      for (int k = 0; k < cnt; ++k) {
        const int fl = (int)aL[k] | ((int)aR[k] << 1);
        if (!fl) continue;
        const float jw = jb[k][2], jh = jb[k][3];
        const float jx1 = jb[k][0] - 0.5f * jw, jy1 = jb[k][1] - 0.5f * jh;
        const float jx2 = jb[k][0] + 0.5f * jw, jy2 = jb[k][1] + 0.5f * jh;
        const float iw = fminf(x2, jx2) - fmaxf(x1, jx1);
        const float ih = fminf(y2, jy2) - fmaxf(y1, jy1);
        const float inter = fmaxf(iw, 0.f) * fmaxf(ih, 0.f);
        const float uni = ai + jw * jh - inter;
        if (inter > 0.5f * uni) {       // iou > NMS_THR (union > 0 always)
          if (fl & 1) hasL = true;
          if (fl & 2) hasR = true;
        }
      }
    }
    __syncthreads();
  }
  if (vi) {
    const long base = ((long)b * NQ + i) * 2;
    outMask[base + 0] = (lf && !hasL) ? 1.0f : 0.0f;
    outMask[base + 1] = (rf && !hasR) ? 1.0f : 0.0f;
  }
}

// ---------------------------------------------------------------------------
// WMMA fragment loaders (contiguous k order k = kb + hi*16 + i, same for A and B, wave32)
// ---------------------------------------------------------------------------
// A (16x32 bf16): lane L holds row M=L%16; hi=L/16.
//   elems i = K = kb + hi*16 + i
__device__ __forceinline__ bf16x16 load_fragA(const bf16_t* buf, int m, int kb, int hi) {
  union { bf16x16 v; bf16x8 h[2]; } u;
  const bf16_t* p = buf + m * HID + kb + hi * 16;
  u.h[0] = *(const bf16x8*)p;
  u.h[1] = *(const bf16x8*)(p + 8);
  return u.v;
}
// B (32x16 bf16): lane L holds col N=L%16; hi=L/16.
//   elems i = K = kb + hi*16 + i  -> contiguous in transposed weight WT[n][k]
__device__ __forceinline__ bf16x16 load_fragB(const bf16_t* WT, int col, int kb, int hi) {
  union { bf16x16 v; bf16x8 h[2]; } u;
  const bf16_t* p = WT + col * HID + kb + hi * 16;
  u.h[0] = *(const bf16x8*)p;
  u.h[1] = *(const bf16x8*)(p + 8);
  return u.v;
}

// One 256->256 GEMM stage for a 32-row tile, distributed over 8 waves.
// wave w: mt = w&1 (M tile), ng = w>>1 (group of 4 N tiles).
__device__ __forceinline__ void gemm_stage(const bf16_t* __restrict__ src,
                                           const bf16_t* __restrict__ WT,
                                           const float* __restrict__ bias,
                                           bf16_t* __restrict__ dst,
                                           bool relu, int mt, int ng, int lane) {
  const int hi = lane >> 4;
  const int ln = lane & 15;
  v8f acc[4] = {};
#pragma unroll
  for (int kb = 0; kb < HID; kb += 32) {
    const bf16x16 a = load_fragA(src, mt * 16 + ln, kb, hi);
#pragma unroll
    for (int ntl = 0; ntl < 4; ++ntl) {
      const int col = (ng * 4 + ntl) * 16 + ln;
      const bf16x16 bfr = load_fragB(WT, col, kb, hi);
      acc[ntl] = __builtin_amdgcn_wmma_f32_16x16x32_bf16(
          false, a, false, bfr, (short)0, acc[ntl], false, false);
    }
  }
#pragma unroll
  for (int ntl = 0; ntl < 4; ++ntl) {
    const int col = (ng * 4 + ntl) * 16 + ln;
    const float bv = bias[col];
#pragma unroll
    for (int r = 0; r < 8; ++r) {
      float v = acc[ntl][r] + bv;
      if (relu) v = fmaxf(v, 0.0f);
      dst[(mt * 16 + r + hi * 8) * HID + col] = (bf16_t)v;
    }
  }
}

// ---------------------------------------------------------------------------
// Kernel 2: fused projector + LN + ReLU + refiner MLP + score + box epilogue
// grid = 64000/32 = 2000 blocks, 256 threads (8 wave32).
// ---------------------------------------------------------------------------
__global__ __launch_bounds__(256) void fused_chain(
    const float* __restrict__ features, const float* __restrict__ boxes,
    const float* __restrict__ maskArr,
    const bf16_t* __restrict__ WT_p1, const float* __restrict__ b_p1,
    const float* __restrict__ ln_g, const float* __restrict__ ln_b,
    const bf16_t* __restrict__ WT_p2, const float* __restrict__ b_p2,
    const bf16_t* __restrict__ WT_b1, const float* __restrict__ b_b1,
    const bf16_t* __restrict__ WT_b2, const float* __restrict__ b_b2,
    const float* __restrict__ W_b3, const float* __restrict__ b_b3,
    const float* __restrict__ W_s, const float* __restrict__ b_s,
    float* __restrict__ out) {
  __shared__ __align__(16) bf16_t bufA[ROWS * HID];   // 16 KB
  __shared__ __align__(16) bf16_t bufB[ROWS * HID];   // 16 KB
  __shared__ float muS[ROWS];
  __shared__ float rsS[ROWS];
  __shared__ float scoreS[ROWS];
  __shared__ float deltaS[ROWS * 4];

  const int t    = threadIdx.x;
  const int lane = t & 31;
  const int w    = t >> 5;
  const int mt   = w & 1;
  const int ng   = w >> 1;
  const long rowBase = (long)blockIdx.x * ROWS;

  // --- load features tile -> bf16 in bufA
  for (int idx = t; idx < ROWS * HID; idx += 256)
    bufA[idx] = (bf16_t)features[rowBase * HID + idx];
  __syncthreads();

  // --- stage 1: h = x @ W_p1 + b_p1  -> bufB
  gemm_stage(bufA, WT_p1, b_p1, bufB, false, mt, ng, lane);
  __syncthreads();

  // --- LayerNorm stats (one thread per row)
  if (t < ROWS) {
    float s = 0.f, ss = 0.f;
    for (int c = 0; c < HID; ++c) {
      const float v = (float)bufB[t * HID + c];
      s += v; ss += v * v;
    }
    const float mu = s * (1.0f / HID);
    const float var = ss * (1.0f / HID) - mu * mu;
    muS[t] = mu;
    rsS[t] = rsqrtf(var + 1e-5f);
  }
  __syncthreads();
  // --- apply LN + ReLU -> bufA
  for (int idx = t; idx < ROWS * HID; idx += 256) {
    const int r = idx >> 8, c = idx & 255;
    float v = (float)bufB[idx];
    v = (v - muS[r]) * rsS[r] * ln_g[c] + ln_b[c];
    bufA[idx] = (bf16_t)fmaxf(v, 0.0f);
  }
  __syncthreads();

  // --- stage 2: ary = h @ W_p2 + b_p2 -> bufB
  gemm_stage(bufA, WT_p2, b_p2, bufB, false, mt, ng, lane);
  __syncthreads();

  // --- scores (reads bufB) in parallel with stage 3 (reads bufB, writes bufA)
  if (t < ROWS) {
    float s = b_s[0];
    for (int c = 0; c < HID; ++c) s += (float)bufB[t * HID + c] * W_s[c];
    scoreS[t] = s;
  }
  // --- stage 3: m1 = relu(ary @ W_b1 + b_b1) -> bufA
  gemm_stage(bufB, WT_b1, b_b1, bufA, true, mt, ng, lane);
  __syncthreads();

  // --- stage 4: m2 = relu(m1 @ W_b2 + b_b2) -> bufB
  gemm_stage(bufA, WT_b2, b_b2, bufB, true, mt, ng, lane);
  __syncthreads();

  // --- stage 5: delta = m2 @ W_b3 + b_b3 (N=4, scalar)
  if (t < ROWS * 4) {
    const int r = t >> 2, c = t & 3;
    float s = b_b3[c];
    for (int k = 0; k < HID; ++k) s += (float)bufB[r * HID + k] * W_b3[k * 4 + c];
    deltaS[t] = s;
  }
  __syncthreads();

  // --- epilogue: refined boxes + score, masked, both sides
  if (t < ROWS * 2) {
    const int r = t >> 1;
    const int side = t & 1;
    const long gr = rowBase + r;              // flattened (b, q)
    const float* bx = boxes + gr * 4;
    const float cx = bx[0], cy = bx[1], wd = bx[2], ht = bx[3];
    float cand[4];
    cand[0] = cx + (side ? 0.1f * wd : 0.0f);
    cand[1] = cy + 0.5f * ht;                 // POS_OFF_Y
    cand[2] = wd * 0.8f;                      // SIZE_RATIO
    cand[3] = ht * 0.8f;
    const float mask = maskArr[gr * 2 + side];
    float* o = out + (gr * 2 + side) * 5;
#pragma unroll
    for (int e = 0; e < 4; ++e)
      o[e] = sigmf(deltaS[r * 4 + e] + inv_sigm(cand[e])) * mask;
    o[4] = scoreS[r] * mask;
  }
}

// ---------------------------------------------------------------------------
// Launcher
// ---------------------------------------------------------------------------
extern "C" void kernel_launch(void* const* d_in, const int* in_sizes, int n_in,
                              void* d_out, int out_size, void* d_ws, size_t ws_size,
                              hipStream_t stream) {
  (void)in_sizes; (void)n_in; (void)out_size; (void)ws_size;
  const float* logits   = (const float*)d_in[0];
  const float* boxes    = (const float*)d_in[1];
  const float* features = (const float*)d_in[2];
  // d_in[3] = memory (unused by the reference)
  const float* W_p1 = (const float*)d_in[4];
  const float* b_p1 = (const float*)d_in[5];
  const float* ln_g = (const float*)d_in[6];
  const float* ln_b = (const float*)d_in[7];
  const float* W_p2 = (const float*)d_in[8];
  const float* b_p2 = (const float*)d_in[9];
  const float* W_b1 = (const float*)d_in[10];
  const float* b_b1 = (const float*)d_in[11];
  const float* W_b2 = (const float*)d_in[12];
  const float* b_b2 = (const float*)d_in[13];
  const float* W_b3 = (const float*)d_in[14];
  const float* b_b3 = (const float*)d_in[15];
  const float* W_s  = (const float*)d_in[16];
  const float* b_s  = (const float*)d_in[17];

  float* out = (float*)d_out;
  float* maskOut = out + (long)BQ * 2 * 5;   // second tuple element: cand_mask

  bf16_t* wt = (bf16_t*)d_ws;
  bf16_t* WT_p1 = wt;
  bf16_t* WT_p2 = wt + 65536;
  bf16_t* WT_b1 = wt + 131072;
  bf16_t* WT_b2 = wt + 196608;

  convert_transpose<<<HID, HID, 0, stream>>>(W_p1, WT_p1);
  convert_transpose<<<HID, HID, 0, stream>>>(W_p2, WT_p2);
  convert_transpose<<<HID, HID, 0, stream>>>(W_b1, WT_b1);
  convert_transpose<<<HID, HID, 0, stream>>>(W_b2, WT_b2);

  masks_kernel<<<dim3(4, NB), 256, 0, stream>>>(logits, boxes, maskOut);

  fused_chain<<<BQ / ROWS, 256, 0, stream>>>(
      features, boxes, maskOut,
      WT_p1, b_p1, ln_g, ln_b, WT_p2, b_p2,
      WT_b1, b_b1, WT_b2, b_b2, W_b3, b_b3, W_s, b_s, out);
}
